// AMRSRN_34385508172358
// MI455X (gfx1250) — matrix operand order repaired and checked by reading, not compiled, not run on hardware
//
#include <hip/hip_runtime.h>
#include <hip/hip_bf16.h>

typedef __attribute__((ext_vector_type(16))) _Float16 v16h;
typedef __attribute__((ext_vector_type(8)))  _Float16 v8h;
typedef __attribute__((ext_vector_type(8)))  float    v8f;

namespace {
constexpr int NGRIDS = 64;
constexpr int VOX    = 64 * 64 * 64;   // 64^3 voxels per channel
constexpr int LPE    = 6;
constexpr int DIN    = 164;            // 36 PE + 128 grid feats
constexpr int KPAD   = 192;            // DIN padded to multiple of 32
constexpr int XSTR   = 200;            // LDS row stride (halves): 400B=100 dwords, 100%64=36 -> conflict-free rows, 16B aligned
constexpr int YSTR   = 72;             // 144B=36 dwords, same property
constexpr int NODES  = 64;
constexpr int TILE   = 64;             // points per block
constexpr int NTHR   = 256;            // 8 wave32
}

__device__ __forceinline__ float snakef(float v) {
  float s = __sinf(v);
  return 0.5f * v + s * s;
}

// A fragment (16x32 f16): lanes 0-15 row M=lane, halves 0..7 = K kb+0..7, 8..15 = K kb+16..23;
// lanes 16-31 same rows, K offset +8.
__device__ __forceinline__ v16h frag_a(const _Float16* row, int kb, int lane) {
  const int kh = (lane >> 4) << 3;
  v8h lo = *(const v8h*)(row + kb + kh);
  v8h hi = *(const v8h*)(row + kb + 16 + kh);
  v16h a;
#pragma unroll
  for (int i = 0; i < 8; ++i) { a[i] = lo[i]; a[i + 8] = hi[i]; }
  return a;
}

// B fragment (32x16 f16): lane%16 = column N, halves 0..7 = K kb+0..7, 8..15 = K kb+16..23 (+8 for upper half-wave).
__device__ __forceinline__ v16h frag_b(const _Float16* col, int kb, int lane) {
  const _Float16* p = col + kb + ((lane >> 4) << 3);
  v8h lo = *(const v8h*)(p);
  v8h hi = *(const v8h*)(p + 16);
  v16h b;
#pragma unroll
  for (int i = 0; i < 8; ++i) { b[i] = lo[i]; b[i + 8] = hi[i]; }
  return b;
}

__global__ __launch_bounds__(NTHR, 2) void amrsrn_fused_kernel(
    const float* __restrict__ xg,
    const float* __restrict__ gscale,
    const float* __restrict__ gtrans,
    const float* __restrict__ fgrid,
    const float* __restrict__ W0g, const float* __restrict__ b0g,
    const float* __restrict__ W1g, const float* __restrict__ b1g,
    const float* __restrict__ W2g, const float* __restrict__ b2g,
    float* __restrict__ out, int npts)
{
  // ~62KB static LDS, with phase-aliased reuse (barrier-protected).
  __shared__ __align__(16) _Float16 Xs [TILE * XSTR];   // feature tile; later aliased as Y2
  __shared__ __align__(16) _Float16 W0T[NODES * XSTR];  // W0^T (f16, K-padded); later aliased as Y1
  __shared__ __align__(16) _Float16 W1T[NODES * YSTR];  // W1^T (f16, K-padded to 72)
  __shared__ float xsh [TILE * 3];
  __shared__ float gscl[NGRIDS * 3];
  __shared__ float gtrn[NGRIDS * 3];
  __shared__ float b0s[NODES], b1s[NODES], w2s[NODES];
  __shared__ float b2s;

  const int tid  = threadIdx.x;
  const int lane = tid & 31;
  const int wv   = tid >> 5;               // wave id 0..7
  const int p0   = blockIdx.x * TILE;

  // ---------- phase 0: stage weights & params ----------
  for (int t = tid; t < NODES * XSTR; t += NTHR) {
    int n = t / XSTR, k = t - n * XSTR;
    float v = (k < DIN) ? W0g[k * NODES + n] : 0.0f;
    W0T[n * XSTR + k] = (_Float16)v;
  }
  for (int t = tid; t < NODES * YSTR; t += NTHR) {
    int n = t / YSTR, k = t - n * YSTR;
    float v = (k < NODES) ? W1g[k * NODES + n] : 0.0f;
    W1T[n * YSTR + k] = (_Float16)v;
  }
  if (tid < NODES) { b0s[tid] = b0g[tid]; b1s[tid] = b1g[tid]; w2s[tid] = W2g[tid]; }
  if (tid == 0) b2s = b2g[0];
  for (int t = tid; t < NGRIDS * 3; t += NTHR) { gscl[t] = gscale[t]; gtrn[t] = gtrans[t]; }
  for (int t = tid; t < TILE * 3; t += NTHR) {
    int gi = p0 * 3 + t;
    xsh[t] = (gi < npts * 3) ? xg[gi] : 0.0f;
  }
  __syncthreads();

  // ---------- phase 1: positional encoding + zero K-pad ----------
  for (int t = tid; t < TILE * LPE * 3; t += NTHR) {
    int p = t / 18, j = t - p * 18;
    int l = j / 3, d = j - l * 3;
    float ang = xsh[p * 3 + d] * (3.14159265358979323846f * (float)(1 << l));
    Xs[p * XSTR + j]      = (_Float16)__sinf(ang);
    Xs[p * XSTR + 18 + j] = (_Float16)__cosf(ang);
  }
  for (int t = tid; t < TILE * (XSTR - DIN); t += NTHR) {
    int p = t / (XSTR - DIN), k = DIN + (t - p * (XSTR - DIN));
    Xs[p * XSTR + k] = (_Float16)0.0f;
  }

  // ---------- phase 2: 64-grid trilinear gather (L2-resident, gather-bound) ----------
  for (int t = tid; t < TILE * NGRIDS; t += NTHR) {
    int p = t & (TILE - 1);
    int g = t >> 6;
    float px = xsh[p * 3 + 0], py = xsh[p * 3 + 1], pz = xsh[p * 3 + 2];
    float tx = px * gscl[g * 3 + 0] + gtrn[g * 3 + 0];
    float ty = py * gscl[g * 3 + 1] + gtrn[g * 3 + 1];
    float tz = pz * gscl[g * 3 + 2] + gtrn[g * 3 + 2];
    float ix = (tx + 1.0f) * 0.5f * 63.0f;
    float iy = (ty + 1.0f) * 0.5f * 63.0f;
    float iz = (tz + 1.0f) * 0.5f * 63.0f;
    float fx = floorf(ix), fy = floorf(iy), fz = floorf(iz);
    float wx = ix - fx, wy = iy - fy, wz = iz - fz;
    int x0 = (int)fx, y0 = (int)fy, z0 = (int)fz;
    const float* vol = fgrid + (size_t)g * (2 * VOX);
    float a0 = 0.0f, a1 = 0.0f;
#pragma unroll
    for (int dz = 0; dz < 2; ++dz) {
      int zi = z0 + dz;  bool vz = (unsigned)zi < 64u;
      int zc = min(max(zi, 0), 63);
      float wz_ = dz ? wz : 1.0f - wz;
#pragma unroll
      for (int dy = 0; dy < 2; ++dy) {
        int yi = y0 + dy;  bool vy = (unsigned)yi < 64u;
        int yc = min(max(yi, 0), 63);
        float wy_ = dy ? wy : 1.0f - wy;
#pragma unroll
        for (int dx = 0; dx < 2; ++dx) {
          int xi = x0 + dx;  bool vx = (unsigned)xi < 64u;
          int xc = min(max(xi, 0), 63);
          float wx_ = dx ? wx : 1.0f - wx;
          if (vz && vy && vx) {
            int idx = (zc * 64 + yc) * 64 + xc;
            float w = wz_ * wy_ * wx_;
            a0 += w * vol[idx];
            a1 += w * vol[VOX + idx];
          }
        }
      }
    }
    Xs[p * XSTR + 36 + 2 * g]     = (_Float16)a0;
    Xs[p * XSTR + 36 + 2 * g + 1] = (_Float16)a1;
  }
  __syncthreads();

  // ---------- phase 3: GEMM0  [64x192] @ [192x64]  via v_wmma_f32_16x16x32_f16 ----------
  const int mtile  = wv >> 1;                 // 0..3
  const int ntbase = (wv & 1) << 1;           // 0 or 2
  const int mrow   = mtile * 16 + (lane & 15);
  const _Float16* arow0 = &Xs[mrow * XSTR];
  const _Float16* bcol0 = &W0T[(ntbase * 16 + (lane & 15)) * XSTR];
  const _Float16* bcol1 = &W0T[((ntbase + 1) * 16 + (lane & 15)) * XSTR];
  v8f acc0 = {}, acc1 = {};
#pragma unroll
  for (int kb = 0; kb < KPAD; kb += 32) {
    v16h a  = frag_a(arow0, kb, lane);
    v16h bA = frag_b(bcol0, kb, lane);
    v16h bB = frag_b(bcol1, kb, lane);
    acc0 = __builtin_amdgcn_wmma_f32_16x16x32_f16(false, a, false, bA, (short)0, acc0, false, false);
    acc1 = __builtin_amdgcn_wmma_f32_16x16x32_f16(false, a, false, bB, (short)0, acc1, false, false);
  }
  __syncthreads();   // all waves done reading W0T -> safe to alias as Y1

  _Float16* Y1 = W0T;                         // [TILE x YSTR], alias
  {
    int mb = mtile * 16 + ((lane >> 4) << 3);
    int n0 = ntbase * 16 + (lane & 15);
    int n1 = (ntbase + 1) * 16 + (lane & 15);
#pragma unroll
    for (int r = 0; r < 8; ++r) {
      Y1[(mb + r) * YSTR + n0] = (_Float16)snakef(acc0[r] + b0s[n0]);
      Y1[(mb + r) * YSTR + n1] = (_Float16)snakef(acc1[r] + b0s[n1]);
    }
  }
  __syncthreads();

  // ---------- phase 4: GEMM1  [64x64] @ [64x64] ----------
  const _Float16* arow1 = &Y1[mrow * YSTR];
  const _Float16* ccol0 = &W1T[(ntbase * 16 + (lane & 15)) * YSTR];
  const _Float16* ccol1 = &W1T[((ntbase + 1) * 16 + (lane & 15)) * YSTR];
  v8f d0 = {}, d1 = {};
#pragma unroll
  for (int kb = 0; kb < NODES; kb += 32) {
    v16h a  = frag_a(arow1, kb, lane);
    v16h bA = frag_b(ccol0, kb, lane);
    v16h bB = frag_b(ccol1, kb, lane);
    d0 = __builtin_amdgcn_wmma_f32_16x16x32_f16(false, a, false, bA, (short)0, d0, false, false);
    d1 = __builtin_amdgcn_wmma_f32_16x16x32_f16(false, a, false, bB, (short)0, d1, false, false);
  }

  _Float16* Y2 = Xs;                          // [TILE x YSTR], alias (Xs dead after GEMM0)
  {
    int mb = mtile * 16 + ((lane >> 4) << 3);
    int n0 = ntbase * 16 + (lane & 15);
    int n1 = (ntbase + 1) * 16 + (lane & 15);
#pragma unroll
    for (int r = 0; r < 8; ++r) {
      Y2[(mb + r) * YSTR + n0] = (_Float16)snakef(d0[r] + b1s[n0]);
      Y2[(mb + r) * YSTR + n1] = (_Float16)snakef(d1[r] + b1s[n1]);
    }
  }
  __syncthreads();

  // ---------- phase 5: final 64->1 dot per point ----------
  if (tid < TILE && (p0 + tid) < npts) {
    float acc = b2s;
#pragma unroll 8
    for (int n = 0; n < NODES; ++n)
      acc += (float)Y2[tid * YSTR + n] * w2s[n];
    out[p0 + tid] = acc;
  }
}

extern "C" void kernel_launch(void* const* d_in, const int* in_sizes, int n_in,
                              void* d_out, int out_size, void* d_ws, size_t ws_size,
                              hipStream_t stream) {
  const float* xg     = (const float*)d_in[0];
  const float* gscale = (const float*)d_in[1];
  const float* gtrans = (const float*)d_in[2];
  const float* fgrid  = (const float*)d_in[3];
  const float* W0g    = (const float*)d_in[4];
  const float* b0g    = (const float*)d_in[5];
  const float* W1g    = (const float*)d_in[6];
  const float* b1g    = (const float*)d_in[7];
  const float* W2g    = (const float*)d_in[8];
  const float* b2g    = (const float*)d_in[9];
  float* out = (float*)d_out;

  const int npts   = in_sizes[0] / 3;
  const int blocks = (npts + TILE - 1) / TILE;

  amrsrn_fused_kernel<<<blocks, NTHR, 0, stream>>>(
      xg, gscale, gtrans, fgrid, W0g, b0g, W1g, b1g, W2g, b2g, out, npts);
}
